// MyViTBlock_87385404604988
// MI455X (gfx1250) — matrix operand order repaired and checked by reading, not compiled, not run on hardware
//
#include <hip/hip_runtime.h>
#include <hip/hip_bf16.h>
#include <math.h>

// ---------------------------------------------------------------------------
// MyViTBlock for MI455X (gfx1250): bf16 WMMA everywhere, flash attention,
// fused residuals. Shapes: B=4, S=2048, D=500 (pad 512), H=10, DH=50 (pad 64).
// ---------------------------------------------------------------------------

typedef __bf16 bf16;
typedef __attribute__((ext_vector_type(16))) __bf16 v16bf;
typedef __attribute__((ext_vector_type(8)))  __bf16 v8bf;
typedef __attribute__((ext_vector_type(8)))  float  v8f;

#define NB    4
#define NS    2048
#define ND    500
#define NH    10
#define NDH   50
#define NBS   (NB*NS)      // 8192 rows
#define DHP   64           // padded head dim
#define DPAD  512          // padded model dim
#define A1STR (NH*DHP)     // 640: ln1 output row stride (head-padded)
#define LNEPS 1e-5f

union Frag { v16bf v; v8bf h[2]; };

__device__ __forceinline__ bf16 f2bf(float x){ return (bf16)x; }

__device__ __forceinline__ v8f wmma_bf16(v16bf a, v16bf b, v8f c){
  return __builtin_amdgcn_wmma_f32_16x16x32_bf16(false, a, false, b, (short)0, c,
                                                 false, false);
}

// A-fragment (16x32 bf16) from row-major buffer. p = row pointer at K-chunk
// start (row selected per-lane by caller). Order used here (not ISA table order): lane L holds M=L%16,
// element j -> K = j + (L>=16?16:0)  => one contiguous 16-run.
__device__ __forceinline__ v16bf load_a_frag(const bf16* p, int lane){
  Frag u;
  const int c0 = (lane & 16) ? 16 : 0;
  u.h[0] = *(const v8bf*)(p + c0);
  u.h[1] = *(const v8bf*)(p + c0 + 8);
  return u.v;
}
// B-fragment (32x16 bf16): lane L holds N, element j -> K = j + (L>=16?16:0).
// Caller passes pointer to (row N, K base incl. lane-half offset): one 32B load.

// ---------------------------------------------------------------------------
// Weight padding / conversion
// ---------------------------------------------------------------------------
__global__ __launch_bounds__(256) void pad_qkv_w_kernel(
    const float* __restrict__ Wq, const float* __restrict__ Wk,
    const float* __restrict__ Wv, bf16* __restrict__ dst){
  int idx = blockIdx.x*256 + threadIdx.x;
  const int per = NH*DHP*DHP;                 // 40960
  if (idx >= 3*per) return;
  int mat = idx / per, rem = idx % per;
  int h = rem / (DHP*DHP), e = (rem / DHP) % DHP, d = rem % DHP;
  const float* src = (mat==0) ? Wq : (mat==1 ? Wk : Wv);
  float v = (e < NDH && d < NDH) ? src[(h*NDH + e)*NDH + d] : 0.f;
  dst[idx] = f2bf(v);
}

__global__ __launch_bounds__(256) void pad_ffn_w_kernel(
    const float* __restrict__ W, bf16* __restrict__ dst){
  int idx = blockIdx.x*256 + threadIdx.x;
  if (idx >= DPAD*DPAD) return;
  int e = idx / DPAD, d = idx % DPAD;
  dst[idx] = f2bf((e < ND && d < ND) ? W[e*ND + d] : 0.f);
}

// ---------------------------------------------------------------------------
// LayerNorm 1: x [BS,500] f32 -> a1 [BS, H*64] bf16 (head-padded, pads zeroed)
// ---------------------------------------------------------------------------
__global__ __launch_bounds__(256) void ln1_kernel(
    const float* __restrict__ x, const float* __restrict__ g,
    const float* __restrict__ b, bf16* __restrict__ a1){
  const int row = blockIdx.x, t = threadIdx.x;
  const float* xr = x + (size_t)row*ND;
  float v0 = (t       < ND) ? xr[t]       : 0.f;
  float v1 = (t + 256 < ND) ? xr[t + 256] : 0.f;
  __shared__ float red[256];
  red[t] = v0 + v1; __syncthreads();
  #pragma unroll
  for (int k = 128; k > 0; k >>= 1){ if (t < k) red[t] += red[t+k]; __syncthreads(); }
  const float mu = red[0] * (1.f/ND); __syncthreads();
  float d0 = v0 - mu, d1 = v1 - mu;
  red[t] = ((t < ND) ? d0*d0 : 0.f) + ((t+256 < ND) ? d1*d1 : 0.f); __syncthreads();
  #pragma unroll
  for (int k = 128; k > 0; k >>= 1){ if (t < k) red[t] += red[t+k]; __syncthreads(); }
  const float rstd = rsqrtf(red[0] * (1.f/ND) + LNEPS);
  bf16* ar = a1 + (size_t)row*A1STR;
  if (t < ND)       ar[(t/NDH)*DHP + (t%NDH)] = f2bf(d0*rstd*g[t] + b[t]);
  if (t + 256 < ND){ int u = t+256; ar[(u/NDH)*DHP + (u%NDH)] = f2bf(d1*rstd*g[u] + b[u]); }
  for (int p = t; p < NH*(DHP-NDH); p += 256){       // zero the 14 pad cols/head
    int h = p / (DHP-NDH), j = NDH + p % (DHP-NDH);
    ar[h*DHP + j] = (bf16)0.f;
  }
}

// ---------------------------------------------------------------------------
// LayerNorm 2: in [BS,512] f32 (cols<500 valid) -> a2 [BS,512] bf16 (pads zero)
// ---------------------------------------------------------------------------
__global__ __launch_bounds__(256) void ln2_kernel(
    const float* __restrict__ in, const float* __restrict__ g,
    const float* __restrict__ b, bf16* __restrict__ a2){
  const int row = blockIdx.x, t = threadIdx.x;
  const float* xr = in + (size_t)row*DPAD;
  float v0 = (t       < ND) ? xr[t]       : 0.f;
  float v1 = (t + 256 < ND) ? xr[t + 256] : 0.f;
  __shared__ float red[256];
  red[t] = v0 + v1; __syncthreads();
  #pragma unroll
  for (int k = 128; k > 0; k >>= 1){ if (t < k) red[t] += red[t+k]; __syncthreads(); }
  const float mu = red[0] * (1.f/ND); __syncthreads();
  float d0 = v0 - mu, d1 = v1 - mu;
  red[t] = ((t < ND) ? d0*d0 : 0.f) + ((t+256 < ND) ? d1*d1 : 0.f); __syncthreads();
  #pragma unroll
  for (int k = 128; k > 0; k >>= 1){ if (t < k) red[t] += red[t+k]; __syncthreads(); }
  const float rstd = rsqrtf(red[0] * (1.f/ND) + LNEPS);
  bf16* ar = a2 + (size_t)row*DPAD;
  if (t < ND)        ar[t]     = f2bf(d0*rstd*g[t] + b[t]);
  if (t + 256 < ND){ int u=t+256; ar[u] = f2bf(d1*rstd*g[u] + b[u]); }
  if (t < DPAD - ND) ar[ND + t] = (bf16)0.f;         // 12 pad cols
}

// ---------------------------------------------------------------------------
// QKV projection: per-head [BS,64]x[64,64] via WMMA.
// a1 [BS,640] bf16; Wp [3][H][64][64] bf16 (y = x @ W^T => B frag (N=e,K=d)=W[e][d]).
// Q,K stored [B,H,S,64]; V stored transposed Vt [B,H,64,S] (packed epilogue store).
// grid (BS/128, H), block 256 (8 waves x 16 rows).
// ---------------------------------------------------------------------------
__global__ __launch_bounds__(256) void qkv_kernel(
    const bf16* __restrict__ a1, const bf16* __restrict__ wq,
    const bf16* __restrict__ wk, const bf16* __restrict__ wv,
    const float* __restrict__ bq, const float* __restrict__ bk,
    const float* __restrict__ bv,
    bf16* __restrict__ Qd, bf16* __restrict__ Kd, bf16* __restrict__ Vtd){
  const int w = threadIdx.x >> 5, lane = threadIdx.x & 31;
  const int h = blockIdx.y;
  const int m0 = blockIdx.x*128 + w*16;
  const int bIdx = m0 / NS, s0l = m0 % NS;
  const int e_ = lane & 15;
  const int kb = (lane & 16) ? 16 : 0;
  const int roff = (lane & 16) ? 8 : 0;

  const bf16* arow = a1 + (size_t)(m0 + (lane & 15))*A1STR + h*DHP;
  const v16bf af0  = load_a_frag(arow,      lane);
  const v16bf af32 = load_a_frag(arow + 32, lane);

  const bf16* Wm[3] = { wq + (size_t)h*DHP*DHP, wk + (size_t)h*DHP*DHP,
                        wv + (size_t)h*DHP*DHP };
  const float* Bm[3] = { bq + h*NDH, bk + h*NDH, bv + h*NDH };
  bf16* Om[2] = { Qd, Kd };

  #pragma unroll
  for (int mat = 0; mat < 3; ++mat){
    #pragma unroll
    for (int nt = 0; nt < 4; ++nt){
      const int e = nt*16 + e_;
      const bf16* wp = Wm[mat] + (size_t)e*DHP + kb;
      v8f acc = {};
      acc = wmma_bf16(af0,  *(const v16bf*)(wp),      acc);
      acc = wmma_bf16(af32, *(const v16bf*)(wp + 32), acc);
      const float bb = (e < NDH) ? Bm[mat][e] : 0.f;
      if (mat == 2){
        v8bf pk;
        #pragma unroll
        for (int r = 0; r < 8; ++r) pk[r] = f2bf(acc[r] + bb);
        *(v8bf*)(Vtd + ((size_t)(bIdx*NH + h)*DHP + e)*NS + s0l + roff) = pk;
      } else {
        bf16* dst = Om[mat];
        const size_t base = ((size_t)(bIdx*NH + h)*NS + s0l + roff)*DHP + e;
        #pragma unroll
        for (int r = 0; r < 8; ++r) dst[base + (size_t)r*DHP] = f2bf(acc[r] + bb);
      }
    }
  }
}

// ---------------------------------------------------------------------------
// Flash attention per (b,h): out1[b,s,h*50+d] = x[...] + softmax(QK^T/sqrt(50))V.
// grid (S/128, H, B), block 256 (8 waves x 16 query rows). Online softmax,
// 32-key tiles, P transposed C->A layout through LDS (wave-local, s_wait_dscnt).
// ---------------------------------------------------------------------------
__global__ __launch_bounds__(256) void flash_kernel(
    const bf16* __restrict__ Qd, const bf16* __restrict__ Kd,
    const bf16* __restrict__ Vtd, const float* __restrict__ x,
    float* __restrict__ out1){
  __shared__ bf16 pbuf[8*16*32];                    // per-wave 16x32 P tile
  const int w = threadIdx.x >> 5, lane = threadIdx.x & 31;
  const int b = blockIdx.z, h = blockIdx.y;
  const int q0 = blockIdx.x*128 + w*16;
  const size_t bh = (size_t)(b*NH + h);
  const int e_ = lane & 15;
  const int roff = (lane & 16) ? 8 : 0;
  const int kb16 = (lane & 16) ? 16 : 0;
  bf16* pb = pbuf + w*512;

  const bf16* Qrow = Qd + (bh*NS + q0 + e_)*DHP;
  const v16bf qa0  = load_a_frag(Qrow,      lane);
  const v16bf qa32 = load_a_frag(Qrow + 32, lane);

  v8f o0 = {}, o1 = {}, o2 = {}, o3 = {};
  float mi[8], li[8];
  #pragma unroll
  for (int r = 0; r < 8; ++r){ mi[r] = -INFINITY; li[r] = 0.f; }
  const float scale = 0.14142135623730951f;          // 1/sqrt(50)

  for (int t0 = 0; t0 < NS; t0 += 32){
    // --- scores: C[M=q, N=t], B frag (N=t, K=d) = K[t][d]
    v8f s0 = {}, s1 = {};
    {
      const bf16* kp0 = Kd + (bh*NS + t0      + e_)*DHP + kb16;
      const bf16* kp1 = Kd + (bh*NS + t0 + 16 + e_)*DHP + kb16;
      s0 = wmma_bf16(qa0,  *(const v16bf*)(kp0),      s0);
      s0 = wmma_bf16(qa32, *(const v16bf*)(kp0 + 32), s0);
      s1 = wmma_bf16(qa0,  *(const v16bf*)(kp1),      s1);
      s1 = wmma_bf16(qa32, *(const v16bf*)(kp1 + 32), s1);
    }
    float mnew[8], corr[8];
    #pragma unroll
    for (int r = 0; r < 8; ++r){ s0[r] *= scale; s1[r] *= scale; }
    #pragma unroll
    for (int r = 0; r < 8; ++r){
      float v = fmaxf(s0[r], s1[r]);                 // row lives across 16 lanes
      #pragma unroll
      for (int m = 1; m < 16; m <<= 1) v = fmaxf(v, __shfl_xor(v, m, 32));
      mnew[r] = fmaxf(mi[r], v);
      corr[r] = __expf(mi[r] - mnew[r]);
    }
    #pragma unroll
    for (int r = 0; r < 8; ++r){
      s0[r] = __expf(s0[r] - mnew[r]);
      s1[r] = __expf(s1[r] - mnew[r]);
    }
    #pragma unroll
    for (int r = 0; r < 8; ++r){
      float v = s0[r] + s1[r];
      #pragma unroll
      for (int m = 1; m < 16; m <<= 1) v += __shfl_xor(v, m, 32);
      li[r] = li[r]*corr[r] + v;
      mi[r] = mnew[r];
    }
    #pragma unroll
    for (int r = 0; r < 8; ++r){
      o0[r] *= corr[r]; o1[r] *= corr[r]; o2[r] *= corr[r]; o3[r] *= corr[r];
    }
    // --- P: C layout -> LDS (row-major 16x32) -> A fragment
    #pragma unroll
    for (int r = 0; r < 8; ++r){
      pb[(r + roff)*32 + e_]      = f2bf(s0[r]);
      pb[(r + roff)*32 + 16 + e_] = f2bf(s1[r]);
    }
    asm volatile("s_wait_dscnt 0x0" ::: "memory");   // wave-local LDS RAW
    v16bf pa;
    {
      const bf16* pr = pb + e_*32 + ((lane & 16) ? 16 : 0);
      Frag u; u.h[0] = *(const v8bf*)pr; u.h[1] = *(const v8bf*)(pr + 8);
      pa = u.v;
    }
    // --- O += P @ V: B frag (N=d, K=t) = Vt[d][t] (contiguous in t)
    const bf16* vp = Vtd + (bh*DHP + e_)*NS + t0 + kb16;
    o0 = wmma_bf16(pa, *(const v16bf*)(vp),           o0);
    o1 = wmma_bf16(pa, *(const v16bf*)(vp + 16*NS),   o1);
    o2 = wmma_bf16(pa, *(const v16bf*)(vp + 32*NS),   o2);
    o3 = wmma_bf16(pa, *(const v16bf*)(vp + 48*NS),   o3);
  }
  // --- epilogue: normalize, fuse residual x, scatter head slice
  float inv[8];
  #pragma unroll
  for (int r = 0; r < 8; ++r) inv[r] = 1.f / li[r];
  v8f oacc[4] = { o0, o1, o2, o3 };
  #pragma unroll
  for (int nt = 0; nt < 4; ++nt){
    const int d = nt*16 + e_;
    if (d >= NDH) continue;
    const int col = h*NDH + d;
    #pragma unroll
    for (int r = 0; r < 8; ++r){
      const size_t row = (size_t)b*NS + q0 + roff + r;
      out1[row*DPAD + col] = x[row*ND + col] + oacc[nt][r]*inv[r];
    }
  }
}

// ---------------------------------------------------------------------------
// FFN GEMM: out[BS,512] = A[BS,512](bf16) @ W^T + bias.  W bf16 [512 e][512 d].
// grid (BS/128, 512/64), block 256 (8 waves x 16 rows, 4 N-tiles each).
// Weights are 0.5MB bf16 -> fully L2-resident; direct vector fragment loads.
// ---------------------------------------------------------------------------
__global__ __launch_bounds__(256) void gemm_kernel(
    const bf16* __restrict__ A, const bf16* __restrict__ W,
    const float* __restrict__ bias, float* __restrict__ out){
  const int w = threadIdx.x >> 5, lane = threadIdx.x & 31;
  const int m0 = blockIdx.x*128 + w*16;
  const int n0 = blockIdx.y*64;
  const int e_ = lane & 15;
  const int kb = (lane & 16) ? 16 : 0;
  const int roff = (lane & 16) ? 8 : 0;
  const bf16* arow = A + (size_t)(m0 + e_)*DPAD;

  v8f acc[4] = {{}, {}, {}, {}};
  #pragma unroll 4
  for (int k0 = 0; k0 < DPAD; k0 += 32){
    const v16bf af = load_a_frag(arow + k0, lane);
    const bf16* wp = W + (size_t)n0*DPAD + k0 + kb;
    acc[0] = wmma_bf16(af, *(const v16bf*)(wp + (size_t)(e_     )*DPAD), acc[0]);
    acc[1] = wmma_bf16(af, *(const v16bf*)(wp + (size_t)(e_ + 16)*DPAD), acc[1]);
    acc[2] = wmma_bf16(af, *(const v16bf*)(wp + (size_t)(e_ + 32)*DPAD), acc[2]);
    acc[3] = wmma_bf16(af, *(const v16bf*)(wp + (size_t)(e_ + 48)*DPAD), acc[3]);
  }
  #pragma unroll
  for (int nt = 0; nt < 4; ++nt){
    const int e = n0 + nt*16 + e_;
    const float bb = (e < ND) ? bias[e] : 0.f;
    #pragma unroll
    for (int r = 0; r < 8; ++r)
      out[(size_t)(m0 + roff + r)*DPAD + e] = acc[nt][r] + bb;
  }
}

// ---------------------------------------------------------------------------
// Column batch-stats (training-mode BN, biased var): one block per column.
// ---------------------------------------------------------------------------
__global__ __launch_bounds__(256) void colstats_kernel(
    const float* __restrict__ g, float* __restrict__ mu, float* __restrict__ var){
  const int c = blockIdx.x, t = threadIdx.x;
  float s = 0.f, s2 = 0.f;
  for (int r = t; r < NBS; r += 256){
    float v = g[(size_t)r*DPAD + c];
    s += v; s2 += v*v;
  }
  __shared__ float rs[256], rq[256];
  rs[t] = s; rq[t] = s2; __syncthreads();
  #pragma unroll
  for (int k = 128; k > 0; k >>= 1){
    if (t < k){ rs[t] += rs[t+k]; rq[t] += rq[t+k]; } __syncthreads();
  }
  if (t == 0){
    float m = rs[0] * (1.f/NBS);
    mu[c]  = m;
    var[c] = rq[0] * (1.f/NBS) - m*m;
  }
}

// ---------------------------------------------------------------------------
// BN + ReLU + residual. mode 0: out2 f32 [*,512] + bf16 restage (pads zeroed).
// mode 1: final write to d_out f32 [*,500].
// ---------------------------------------------------------------------------
__global__ __launch_bounds__(256) void bn_relu_add_kernel(
    const float* __restrict__ g, const float* __restrict__ mu,
    const float* __restrict__ var, const float* __restrict__ gamma,
    const float* __restrict__ beta, const float* __restrict__ res,
    float* __restrict__ outf, bf16* __restrict__ outb, int final_mode){
  const int idx = blockIdx.x*256 + threadIdx.x;
  if (idx >= NBS*DPAD) return;
  const int row = idx / DPAD, c = idx % DPAD;
  if (c < ND){
    float v = g[idx];
    float y = (v - mu[c]) * rsqrtf(var[c] + LNEPS) * gamma[c] + beta[c];
    y = fmaxf(y, 0.f);
    float o = res[(size_t)row*DPAD + c] + y;
    if (final_mode) outf[(size_t)row*ND + c] = o;
    else { outf[idx] = o; outb[idx] = f2bf(o); }
  } else if (!final_mode){
    outb[idx] = (bf16)0.f;
  }
}

// ---------------------------------------------------------------------------
// Host-side orchestration
// ---------------------------------------------------------------------------
extern "C" void kernel_launch(void* const* d_in, const int* in_sizes, int n_in,
                              void* d_out, int out_size, void* d_ws, size_t ws_size,
                              hipStream_t stream) {
  (void)in_sizes; (void)n_in; (void)out_size; (void)ws_size;
  const float* x    = (const float*)d_in[0];
  const float* ln1g = (const float*)d_in[1];
  const float* ln1b = (const float*)d_in[2];
  const float* Wq   = (const float*)d_in[3];
  const float* bq   = (const float*)d_in[4];
  const float* Wk   = (const float*)d_in[5];
  const float* bk   = (const float*)d_in[6];
  const float* Wv   = (const float*)d_in[7];
  const float* bv   = (const float*)d_in[8];
  const float* ln2g = (const float*)d_in[9];
  const float* ln2b = (const float*)d_in[10];
  const float* W1   = (const float*)d_in[11];
  const float* b1   = (const float*)d_in[12];
  const float* bn1g = (const float*)d_in[13];
  const float* bn1b = (const float*)d_in[14];
  const float* W2   = (const float*)d_in[15];
  const float* b2   = (const float*)d_in[16];
  const float* bn2g = (const float*)d_in[17];
  const float* bn2b = (const float*)d_in[18];

  // Workspace layout (bytes). Attention-phase buffers (a1,Q,K,Vt = 41,943,040 B)
  // are exactly reused for the FFN phase (a2 8,388,608 + g 16,777,216 + out2
  // 16,777,216 = 41,943,040 B).
  char* ws = (char*)d_ws;
  constexpr size_t OFF_QKVW = 0;                       // 3*10*64*64*2 = 245760
  constexpr size_t OFF_W1P  = OFF_QKVW + 245760;       // 512*512*2    = 524288
  constexpr size_t OFF_W2P  = OFF_W1P  + 524288;
  constexpr size_t OFF_STAT = OFF_W2P  + 524288;       // 2*512*4      = 4096
  constexpr size_t OFF_PH   = OFF_STAT + 8192;         // phase region (aligned)
  constexpr size_t SZ_A1    = (size_t)NBS*A1STR*2;     // 10,485,760
  constexpr size_t SZ_QKV   = (size_t)NB*NH*NS*DHP*2;  // 10,485,760
  constexpr size_t OFF_OUT1 = OFF_PH + SZ_A1 + 3*SZ_QKV;

  bf16*  qkvW = (bf16*)(ws + OFF_QKVW);
  bf16*  w1p  = (bf16*)(ws + OFF_W1P);
  bf16*  w2p  = (bf16*)(ws + OFF_W2P);
  float* stat = (float*)(ws + OFF_STAT);               // mu | var
  bf16*  a1   = (bf16*)(ws + OFF_PH);
  bf16*  Qb   = (bf16*)(ws + OFF_PH + SZ_A1);
  bf16*  Kb   = (bf16*)(ws + OFF_PH + SZ_A1 + SZ_QKV);
  bf16*  Vtb  = (bf16*)(ws + OFF_PH + SZ_A1 + 2*SZ_QKV);
  float* out1 = (float*)(ws + OFF_OUT1);
  // FFN-phase overlays (a1/Q/K/Vt dead after flash_kernel):
  bf16*  a2   = (bf16*)(ws + OFF_PH);
  float* gbuf = (float*)(ws + OFF_PH + (size_t)NBS*DPAD*2);
  float* out2 = (float*)(ws + OFF_PH + (size_t)NBS*DPAD*2 + (size_t)NBS*DPAD*4);

  // 1) weight staging (bf16, zero-padded)
  pad_qkv_w_kernel<<<(3*NH*DHP*DHP + 255)/256, 256, 0, stream>>>(Wq, Wk, Wv, qkvW);
  pad_ffn_w_kernel<<<(DPAD*DPAD + 255)/256, 256, 0, stream>>>(W1, w1p);
  pad_ffn_w_kernel<<<(DPAD*DPAD + 255)/256, 256, 0, stream>>>(W2, w2p);

  // 2) attention path
  ln1_kernel<<<NBS, 256, 0, stream>>>(x, ln1g, ln1b, a1);
  qkv_kernel<<<dim3(NBS/128, NH), 256, 0, stream>>>(
      a1, qkvW, qkvW + NH*DHP*DHP, qkvW + 2*NH*DHP*DHP, bq, bk, bv, Qb, Kb, Vtb);
  flash_kernel<<<dim3(NS/128, NH, NB), 256, 0, stream>>>(Qb, Kb, Vtb, x, out1);

  // 3) FFN layer 1 (Linear + BN(batch stats) + ReLU) + residual
  ln2_kernel<<<NBS, 256, 0, stream>>>(out1, ln2g, ln2b, a2);
  gemm_kernel<<<dim3(NBS/128, DPAD/64), 256, 0, stream>>>(a2, w1p, b1, gbuf);
  colstats_kernel<<<DPAD, 256, 0, stream>>>(gbuf, stat, stat + DPAD);
  bn_relu_add_kernel<<<(NBS*DPAD + 255)/256, 256, 0, stream>>>(
      gbuf, stat, stat + DPAD, bn1g, bn1b, out1, out2, a2 /* restage as a3 */, 0);

  // 4) FFN layer 2 + residual -> d_out
  gemm_kernel<<<dim3(NBS/128, DPAD/64), 256, 0, stream>>>(a2, w2p, b2, gbuf);
  colstats_kernel<<<DPAD, 256, 0, stream>>>(gbuf, stat, stat + DPAD);
  bn_relu_add_kernel<<<(NBS*DPAD + 255)/256, 256, 0, stream>>>(
      gbuf, stat, stat + DPAD, bn2g, bn2b, out2, (float*)d_out, nullptr, 1);
}
